// net_18760417149359
// MI455X (gfx1250) — matrix operand order repaired and checked by reading, not compiled, not run on hardware
//
#include <hip/hip_runtime.h>
#include <hip/hip_bf16.h>

typedef __attribute__((ext_vector_type(16))) _Float16 v16h;
typedef __attribute__((ext_vector_type(8)))  float    v8f;

#define BATCH   4
#define NPTS    2048
#define DMODEL  512
#define DFF     1024
#define KNN     20
#define HEADS   4
#define DKH     128
#define BNROWS  (BATCH*NPTS)          // 8192
#define LN_EPS  1e-6f
#define BN_EPS  1e-5f

#define WMMA_F16(A, B, C) \
  __builtin_amdgcn_wmma_f32_16x16x32_f16(false, (A), false, (B), (short)0, (C), false, false)

// LDS offset = low 32 bits of the generic pointer (generic LDS = aperture<<32 | offset)
#define LDS_OFF(p) ((unsigned)(size_t)(void*)(p))

// One wave32 async 16-byte chunk copy: global f16 -> LDS
#define ASYNC_CP16(ldsoff, gptr)                                              \
  asm volatile("global_load_async_to_lds_b128 %0, %1, off"                    \
               :: "v"(ldsoff), "v"(gptr) : "memory")
#define ASYNC_WAIT_ALL()  asm volatile("s_wait_asynccnt 0x0" ::: "memory")
#define ASYNC_WAIT_6()    asm volatile("s_wait_asynccnt 0x6" ::: "memory")

// ---------------------------------------------------------------------------
// WMMA fragment loaders (wave32, v_wmma_f32_16x16x32_f16)
// A (16x32): lane l -> M=l&15 ; kb=(l>>4)*8 ; a[j] K = kb + (j&7) + (j>=8?16:0)
// B (32x16): lane l -> N=l&15 ; kb=(l>>4)*8 ; b[j] K = kb + (j&7) + (j>=8?16:0)
// C (16x16 f32): lane l -> N=l&15, rows M=(l>>4)*8 + v
// ---------------------------------------------------------------------------
__device__ inline v16h load_frag_a_lds(const _Float16* s, int ld) {
  int l = threadIdx.x & 31;
  const _Float16* r = s + (size_t)(l & 15) * ld;
  int kb = (l >> 4) << 3;
  v16h a;
#pragma unroll
  for (int j = 0; j < 16; ++j) {
    int k = kb + (j & 7) + ((j >= 8) ? 16 : 0);
    a[j] = r[k];
  }
  return a;
}

// B fragment from LDS stored n-major [n][ld]: per lane 8 halves at k=8h.. and 8 at k=16+8h.. (h=l>>4).
__device__ inline v16h load_frag_bt_lds(const _Float16* s, int ld) {
  int l = threadIdx.x & 31;
  const _Float16* r = s + (size_t)(l & 15) * ld + (((l >> 4) << 3));
  v16h b;
#pragma unroll
  for (int j = 0; j < 16; ++j) b[j] = r[(j & 7) + ((j >= 8) ? 16 : 0)];
  return b;
}

// ---------------------------------------------------------------------------
// KNN: one thread per query point; points staged in LDS; top-20 by -||d||^2
// ---------------------------------------------------------------------------
__global__ __launch_bounds__(128) void knn_kernel(const float* __restrict__ x,
                                                  int* __restrict__ idx) {
  __shared__ float px[NPTS], py[NPTS], pz[NPTS];
  int b = blockIdx.x / (NPTS / 128);
  int nblk = blockIdx.x % (NPTS / 128);
  const float* xb = x + (size_t)b * 3 * NPTS;
  for (int i = threadIdx.x; i < NPTS; i += 128) {
    px[i] = xb[i]; py[i] = xb[NPTS + i]; pz[i] = xb[2 * NPTS + i];
  }
  __syncthreads();
  int n = nblk * 128 + threadIdx.x;
  float cx = px[n], cy = py[n], cz = pz[n];
  float best[KNN]; int bidx[KNN];
#pragma unroll
  for (int j = 0; j < KNN; ++j) { best[j] = -3.4e38f; bidx[j] = 0; }
  for (int m = 0; m < NPTS; ++m) {
    float dx = px[m] - cx, dy = py[m] - cy, dz = pz[m] - cz;
    float d = -(dx * dx + dy * dy + dz * dz);
    if (d > best[KNN - 1]) {
      int j = KNN - 1;
      while (j > 0 && d > best[j - 1]) {
        best[j] = best[j - 1]; bidx[j] = bidx[j - 1]; --j;
      }
      best[j] = d; bidx[j] = m;
    }
  }
  int* out = idx + ((size_t)b * NPTS + n) * KNN;
#pragma unroll
  for (int j = 0; j < KNN; ++j) out[j] = bidx[j];
}

// ---------------------------------------------------------------------------
// Pad+convert weights to f16: out[n][kp] = W[n][k<kact], zero padded.
// Also used as a plain f32->f16 converter when kact==kp.
// ---------------------------------------------------------------------------
__global__ __launch_bounds__(256) void padcvt_kernel(const float* __restrict__ w,
                                                     int cout, int kact, int kp,
                                                     _Float16* __restrict__ out) {
  int i = blockIdx.x * 256 + threadIdx.x;
  if (i >= cout * kp) return;
  int n = i / kp, k = i - n * kp;
  int kc = (k < kact) ? k : 0;
  float v = w[(size_t)n * kact + kc];
  out[i] = (k < kact) ? (_Float16)v : (_Float16)0.0f;
}

// vT[(b*DMODEL + d)*NPTS + m] = v[(b*NPTS + m)*DMODEL + d]   (f16 -> f16)
__global__ __launch_bounds__(256) void vtrans_kernel(const _Float16* __restrict__ v,
                                                     _Float16* __restrict__ vt) {
  size_t i = (size_t)blockIdx.x * 256 + threadIdx.x;
  int m = (int)(i % NPTS);
  int d = (int)((i / NPTS) % DMODEL);
  int b = (int)(i / ((size_t)NPTS * DMODEL));
  vt[i] = v[((size_t)b * NPTS + m) * DMODEL + d];
}

// ---------------------------------------------------------------------------
// Async-copy nchunks*16B from global f16 image into LDS (64-thread block).
// ---------------------------------------------------------------------------
__device__ inline void stage_async64(const _Float16* __restrict__ g,
                                     _Float16* lds, int nchunks) {
  for (int c = threadIdx.x; c < nchunks; c += 64) {
    unsigned la = LDS_OFF(&lds[(size_t)c << 3]);
    const _Float16* ga = g + ((size_t)c << 3);
    ASYNC_CP16(la, ga);
  }
}

// In-block WMMA conv: H_out[32][cout] = relu(H_in[32][kp] x Wlds[cout][kp]^T)
__device__ inline void block_conv(const _Float16* in, int kp,
                                  const _Float16* wb, int cout,
                                  _Float16* out) {
  int wave = threadIdx.x >> 5;
  int l = threadIdx.x & 31;
  int nn = l & 15, mb = (l >> 4) * 8;
  for (int nt = 0; nt < cout; nt += 16) {
    v8f acc = {0.f, 0.f, 0.f, 0.f, 0.f, 0.f, 0.f, 0.f};
    for (int k0 = 0; k0 < kp; k0 += 32) {
      v16h a = load_frag_a_lds(in + (size_t)(wave * 16) * kp + k0, kp);
      v16h bf = load_frag_bt_lds(wb + (size_t)nt * kp + k0, kp);
      acc = WMMA_F16(a, bf, acc);
    }
#pragma unroll
    for (int v = 0; v < 8; ++v) {
      float val = fmaxf(acc[v], 0.0f);
      out[(size_t)(wave * 16 + mb + v) * cout + (nt + nn)] = (_Float16)val;
    }
  }
}

// ---------------------------------------------------------------------------
// Fused DGCNN: graph feature + conv1..4 + max-over-k, one block per point.
// Weights arrive as pre-padded f16 images, staged into LDS with async loads.
// cat is written in f16 (it only feeds the conv5 GEMM). Block = 64 thr.
// ---------------------------------------------------------------------------
__global__ __launch_bounds__(64) void dgcnn_kernel(
    const float* __restrict__ x, const int* __restrict__ idx,
    const _Float16* __restrict__ c1p, const _Float16* __restrict__ c2p,
    const _Float16* __restrict__ c3p, const _Float16* __restrict__ c4p,
    _Float16* __restrict__ cat) {
  __shared__ __align__(16) _Float16 feat[32 * 32];
  __shared__ __align__(16) _Float16 hA[32 * 64];
  __shared__ __align__(16) _Float16 hB[32 * 64];
  __shared__ __align__(16) _Float16 hC[32 * 128];
  __shared__ __align__(16) _Float16 hD[32 * 256];
  __shared__ __align__(16) _Float16 wbuf[256 * 128];   // 64 KB, reused per conv
  __shared__ float ctr[3];
  __shared__ float nco[KNN][3];
  __shared__ float dst[KNN];

  int bn = blockIdx.x;
  int b = bn / NPTS, n = bn % NPTS;
  const float* xb = x + (size_t)b * 3 * NPTS;
  int tid = threadIdx.x;

  stage_async64(c1p, wbuf, 64 * 32 / 8);     // conv1 weights in flight

  if (tid < 3) ctr[tid] = xb[tid * NPTS + n];
  __syncthreads();
  if (tid < KNN) {
    int m = idx[((size_t)b * NPTS + n) * KNN + tid];
    float nx = xb[m], ny = xb[NPTS + m], nz = xb[2 * NPTS + m];
    nco[tid][0] = nx; nco[tid][1] = ny; nco[tid][2] = nz;
    float dx = nx - ctr[0], dy = ny - ctr[1], dz = nz - ctr[2];
    dst[tid] = sqrtf(dx * dx + dy * dy + dz * dz);
  }
  __syncthreads();
  for (int i = tid; i < 32 * 32; i += 64) {
    int j = i >> 5, c = i & 31;
    float v = 0.0f;
    if (j < KNN) {
      if (c < 3)       v = ctr[c];
      else if (c < 6)  v = nco[j][c - 3];
      else if (c < 26) v = dst[c - 6];
    }
    feat[i] = (_Float16)v;
  }
  ASYNC_WAIT_ALL();
  __syncthreads();

  _Float16* co = cat + (size_t)bn * DMODEL;

  block_conv(feat, 32, wbuf, 64, hA);
  __syncthreads();
  { int c = tid; float mx = 0.f;
    for (int j = 0; j < KNN; ++j) mx = fmaxf(mx, (float)hA[j * 64 + c]);
    co[c] = (_Float16)mx; }
  stage_async64(c2p, wbuf, 64 * 64 / 8);
  ASYNC_WAIT_ALL();
  __syncthreads();

  block_conv(hA, 64, wbuf, 64, hB);
  __syncthreads();
  { int c = tid; float mx = 0.f;
    for (int j = 0; j < KNN; ++j) mx = fmaxf(mx, (float)hB[j * 64 + c]);
    co[64 + c] = (_Float16)mx; }
  stage_async64(c3p, wbuf, 128 * 64 / 8);
  ASYNC_WAIT_ALL();
  __syncthreads();

  block_conv(hB, 64, wbuf, 128, hC);
  __syncthreads();
  for (int c = tid; c < 128; c += 64) {
    float mx = 0.f;
    for (int j = 0; j < KNN; ++j) mx = fmaxf(mx, (float)hC[j * 128 + c]);
    co[128 + c] = (_Float16)mx;
  }
  stage_async64(c4p, wbuf, 256 * 128 / 8);
  ASYNC_WAIT_ALL();
  __syncthreads();

  block_conv(hC, 128, wbuf, 256, hD);
  __syncthreads();
  for (int c = tid; c < 256; c += 64) {
    float mx = 0.f;
    for (int j = 0; j < KNN; ++j) mx = fmaxf(mx, (float)hD[j * 256 + c]);
    co[256 + c] = (_Float16)mx;
  }
}

// ---------------------------------------------------------------------------
// Tiled WMMA GEMM, all-f16 operands, double-buffered async LDS staging.
// C[M,N] = act(scale*(A x B^T) + bias);  A:[M,K] halves, B:[N,K] halves.
// Block tile 128x64, K-step 64; 8 waves (4xM, 2xN); wave = 32x32.
// Per stage each wave issues 6 global_load_async_to_lds_b128 (A:4, B:2),
// then s_wait_asynccnt 6 overlaps next-tile loads with the 8 WMMAs.
// M%128==0, N%64==0, K%64==0 for every call here.
// ---------------------------------------------------------------------------
#define GTK 64
__global__ __launch_bounds__(256) void gemm_kernel(
    const _Float16* __restrict__ A, int lda,
    const _Float16* __restrict__ B, int ldb,
    const float* __restrict__ bias,
    void* __restrict__ Cp, int ldc,
    int Kd, float scale, int relu, int c16) {
  __shared__ __align__(16) _Float16 As[2][128 * GTK];   // 2 x 16 KB
  __shared__ __align__(16) _Float16 Bs[2][64 * GTK];    // 2 x  8 KB
  const int tid = threadIdx.x;
  const int m0 = blockIdx.y * 128;
  const int n0 = blockIdx.x * 64;
  const int wave = tid >> 5;
  const int wm = (wave & 3) * 32;
  const int wn = (wave >> 2) * 32;

  v8f acc00 = {0.f,0.f,0.f,0.f,0.f,0.f,0.f,0.f};
  v8f acc01 = acc00, acc10 = acc00, acc11 = acc00;

  auto issue = [&](int k0, int buf) {
    const _Float16* Ab = A + (size_t)m0 * lda + k0;
#pragma unroll
    for (int c = tid; c < 128 * (GTK / 8); c += 256) {    // 4 chunks / thread
      int m = c >> 3, q = (c & 7) << 3;
      unsigned la = LDS_OFF(&As[buf][m * GTK + q]);
      const _Float16* ga = Ab + (size_t)m * lda + q;
      ASYNC_CP16(la, ga);
    }
    const _Float16* Bb = B + (size_t)n0 * ldb + k0;
#pragma unroll
    for (int c = tid; c < 64 * (GTK / 8); c += 256) {     // 2 chunks / thread
      int nn = c >> 3, q = (c & 7) << 3;
      unsigned la = LDS_OFF(&Bs[buf][nn * GTK + q]);
      const _Float16* ga = Bb + (size_t)nn * ldb + q;
      ASYNC_CP16(la, ga);
    }
  };

  const int nk = Kd / GTK;
  issue(0, 0);
  for (int kt = 0; kt < nk; ++kt) {
    int buf = kt & 1;
    if (kt + 1 < nk) {
      issue((kt + 1) * GTK, buf ^ 1);
      ASYNC_WAIT_6();          // previous stage landed (in-order completion)
    } else {
      ASYNC_WAIT_ALL();
    }
    __syncthreads();
    v16h a00 = load_frag_a_lds(&As[buf][(size_t)(wm + 0)  * GTK + 0 ], GTK);
    v16h a01 = load_frag_a_lds(&As[buf][(size_t)(wm + 0)  * GTK + 32], GTK);
    v16h a10 = load_frag_a_lds(&As[buf][(size_t)(wm + 16) * GTK + 0 ], GTK);
    v16h a11 = load_frag_a_lds(&As[buf][(size_t)(wm + 16) * GTK + 32], GTK);
    v16h b00 = load_frag_bt_lds(&Bs[buf][(size_t)(wn + 0)  * GTK + 0 ], GTK);
    v16h b01 = load_frag_bt_lds(&Bs[buf][(size_t)(wn + 0)  * GTK + 32], GTK);
    v16h b10 = load_frag_bt_lds(&Bs[buf][(size_t)(wn + 16) * GTK + 0 ], GTK);
    v16h b11 = load_frag_bt_lds(&Bs[buf][(size_t)(wn + 16) * GTK + 32], GTK);
    acc00 = WMMA_F16(a00, b00, acc00);
    acc00 = WMMA_F16(a01, b01, acc00);
    acc01 = WMMA_F16(a00, b10, acc01);
    acc01 = WMMA_F16(a01, b11, acc01);
    acc10 = WMMA_F16(a10, b00, acc10);
    acc10 = WMMA_F16(a11, b01, acc10);
    acc11 = WMMA_F16(a10, b10, acc11);
    acc11 = WMMA_F16(a11, b11, acc11);
    __syncthreads();
  }

  int l = tid & 31;
  int nn = l & 15;
  int mb = (l >> 4) * 8;
  auto store_tile = [&](v8f acc, int ms, int ns) {
#pragma unroll
    for (int v = 0; v < 8; ++v) {
      int m = m0 + wm + ms + mb + v;
      int nc = n0 + wn + ns + nn;
      float val = acc[v] * scale + (bias ? bias[nc] : 0.0f);
      if (relu) val = fmaxf(val, 0.0f);
      if (c16) ((_Float16*)Cp)[(size_t)m * ldc + nc] = (_Float16)val;
      else     ((float*)Cp)[(size_t)m * ldc + nc] = val;
    }
  };
  store_tile(acc00, 0, 0);
  store_tile(acc01, 0, 16);
  store_tile(acc10, 16, 0);
  store_tile(acc11, 16, 16);
}

// ---------------------------------------------------------------------------
// LayerNorm (unbiased std, (std+eps)); optional f16 output (GEMM feeds).
// ---------------------------------------------------------------------------
__global__ __launch_bounds__(256) void ln_kernel(const float* __restrict__ x,
                                                 const float* __restrict__ g,
                                                 const float* __restrict__ b,
                                                 void* __restrict__ y, int f16o) {
  __shared__ float rs[256], rq[256];
  int row = blockIdx.x;
  const float* xr = x + (size_t)row * DMODEL;
  float s = 0.f, q = 0.f;
  for (int c = threadIdx.x; c < DMODEL; c += 256) {
    float v = xr[c]; s += v; q += v * v;
  }
  rs[threadIdx.x] = s; rq[threadIdx.x] = q;
  __syncthreads();
  for (int o = 128; o > 0; o >>= 1) {
    if (threadIdx.x < o) {
      rs[threadIdx.x] += rs[threadIdx.x + o];
      rq[threadIdx.x] += rq[threadIdx.x + o];
    }
    __syncthreads();
  }
  float mean = rs[0] / (float)DMODEL;
  float var = (rq[0] - (float)DMODEL * mean * mean) / (float)(DMODEL - 1);
  float inv = 1.0f / (sqrtf(fmaxf(var, 0.0f)) + LN_EPS);
  for (int c = threadIdx.x; c < DMODEL; c += 256) {
    float val = g[c] * (xr[c] - mean) * inv + b[c];
    if (f16o) ((_Float16*)y)[(size_t)row * DMODEL + c] = (_Float16)val;
    else      ((float*)y)[(size_t)row * DMODEL + c] = val;
  }
}

// Row softmax over NPTS f32 scores -> f16 output; one block per row.
__global__ __launch_bounds__(256) void softmax_kernel(float* __restrict__ s,
                                                      _Float16* __restrict__ sh) {
  __shared__ float red[256];
  float* r = s + (size_t)blockIdx.x * NPTS;
  _Float16* rh = sh + (size_t)blockIdx.x * NPTS;
  float mx = -3.4e38f;
  for (int c = threadIdx.x; c < NPTS; c += 256) mx = fmaxf(mx, r[c]);
  red[threadIdx.x] = mx;
  __syncthreads();
  for (int o = 128; o > 0; o >>= 1) {
    if (threadIdx.x < o) red[threadIdx.x] = fmaxf(red[threadIdx.x], red[threadIdx.x + o]);
    __syncthreads();
  }
  mx = red[0];
  __syncthreads();
  float sum = 0.f;
  for (int c = threadIdx.x; c < NPTS; c += 256) {
    float e = expf(r[c] - mx); r[c] = e; sum += e;
  }
  red[threadIdx.x] = sum;
  __syncthreads();
  for (int o = 128; o > 0; o >>= 1) {
    if (threadIdx.x < o) red[threadIdx.x] += red[threadIdx.x + o];
    __syncthreads();
  }
  float inv = 1.0f / red[0];
  for (int c = threadIdx.x; c < NPTS; c += 256) rh[c] = (_Float16)(r[c] * inv);
}

// BatchNorm stats over BNROWS samples for each of DFF channels (ddof=0).
__global__ __launch_bounds__(256) void bn_stats_kernel(const float* __restrict__ h,
                                                       float* __restrict__ mv) {
  __shared__ float rs[256], rq[256];
  int f = blockIdx.x;
  float s = 0.f, q = 0.f;
  for (int r = threadIdx.x; r < BNROWS; r += 256) {
    float v = h[(size_t)r * DFF + f]; s += v; q += v * v;
  }
  rs[threadIdx.x] = s; rq[threadIdx.x] = q;
  __syncthreads();
  for (int o = 128; o > 0; o >>= 1) {
    if (threadIdx.x < o) {
      rs[threadIdx.x] += rs[threadIdx.x + o];
      rq[threadIdx.x] += rq[threadIdx.x + o];
    }
    __syncthreads();
  }
  if (threadIdx.x == 0) {
    float mean = rs[0] / (float)BNROWS;
    mv[f] = mean;
    mv[DFF + f] = rq[0] / (float)BNROWS - mean * mean;
  }
}

// hn (f16, feeds w2 GEMM) = g*(h-mean)*rsqrt(var+eps)+beta
__global__ __launch_bounds__(256) void bn_apply_kernel(const float* __restrict__ h,
                                                       const float* __restrict__ g,
                                                       const float* __restrict__ beta,
                                                       const float* __restrict__ mv,
                                                       _Float16* __restrict__ hn) {
  size_t i = (size_t)blockIdx.x * 256 + threadIdx.x;
  int f = (int)(i % DFF);
  float mean = mv[f], var = mv[DFF + f];
  hn[i] = (_Float16)(g[f] * (h[i] - mean) * rsqrtf(var + BN_EPS) + beta[f]);
}

__global__ __launch_bounds__(256) void add_kernel(const float* __restrict__ a,
                                                  const float* __restrict__ b,
                                                  float* __restrict__ y) {
  size_t i = (size_t)blockIdx.x * 256 + threadIdx.x;
  y[i] = a[i] + b[i];
}

// out[b,c,n] = eT[b,n,c] + pT[b,n,c]   (one cloud per call)
__global__ __launch_bounds__(256) void final_kernel(const float* __restrict__ eT,
                                                    const float* __restrict__ pT,
                                                    float* __restrict__ out) {
  size_t i = (size_t)blockIdx.x * 256 + threadIdx.x;
  int n = (int)(i % NPTS);
  int c = (int)((i / NPTS) % DMODEL);
  int b = (int)(i / ((size_t)NPTS * DMODEL));
  size_t t = ((size_t)b * NPTS + n) * DMODEL + c;
  out[i] = eT[t] + pT[t];
}

// ---------------------------------------------------------------------------
extern "C" void kernel_launch(void* const* d_in, const int* in_sizes, int n_in,
                              void* d_out, int out_size, void* d_ws, size_t ws_size,
                              hipStream_t stream) {
  (void)in_sizes; (void)n_in; (void)out_size; (void)ws_size;
  // Input order: src, tgt, then params flattened in JAX pytree (sorted-key) order.
  const float* src = (const float*)d_in[0];
  const float* tgt = (const float*)d_in[1];
  const float* c1  = (const float*)d_in[2];
  const float* c2  = (const float*)d_in[3];
  const float* c3  = (const float*)d_in[4];
  const float* c4  = (const float*)d_in[5];
  const float* c5  = (const float*)d_in[6];
  const float* bk  = (const float*)d_in[7];
  const float* bo  = (const float*)d_in[8];
  const float* bq  = (const float*)d_in[9];
  const float* bv  = (const float*)d_in[10];
  const float* wk  = (const float*)d_in[11];
  const float* wo  = (const float*)d_in[12];
  const float* wq  = (const float*)d_in[13];
  const float* wv  = (const float*)d_in[14];
  const float* fb1 = (const float*)d_in[15];
  const float* fb2 = (const float*)d_in[16];
  const float* fbeta = (const float*)d_in[17];
  const float* fg  = (const float*)d_in[18];
  const float* fw1 = (const float*)d_in[19];
  const float* fw2 = (const float*)d_in[20];
  const float* ln0a = (const float*)d_in[21];
  const float* ln0b = (const float*)d_in[22];
  const float* ln1a = (const float*)d_in[23];
  const float* ln1b = (const float*)d_in[24];
  const float* lnma = (const float*)d_in[25];
  const float* lnmb = (const float*)d_in[26];
  const float* lnoa = (const float*)d_in[27];
  const float* lnob = (const float*)d_in[28];

  const size_t BND = (size_t)BATCH * NPTS * DMODEL;  // 4,194,304 elements
  char* wsb = (char*)d_ws;
  size_t off = 0;
  auto allocb = [&](size_t bytes) {
    void* p = wsb + off; off += (bytes + 255) & ~(size_t)255; return p;
  };

  int*   idxb  = (int*)allocb((size_t)BNROWS * KNN * 4);
  // f32 buffers
  float* eaT   = (float*)allocb(BND * 4);
  float* ebT   = (float*)allocb(BND * 4);
  float* scores = (float*)allocb((size_t)NPTS * NPTS * 4);
  float* hbuf  = (float*)allocb((size_t)BNROWS * DFF * 4);
  float* projo = (float*)allocb(BND * 4);
  float* x1f   = (float*)allocb(BND * 4);
  float* x2f   = (float*)allocb(BND * 4);
  float* paT   = (float*)allocb(BND * 4);
  float* pbT   = (float*)allocb(BND * 4);
  float* mv    = (float*)allocb(2 * DFF * 4);
  // f16 buffers
  _Float16* cat_h   = (_Float16*)allocb(BND * 2);
  _Float16* xln_h   = (_Float16*)allocb(BND * 2);
  _Float16* memln_h = (_Float16*)allocb(BND * 2);
  _Float16* q_h     = (_Float16*)allocb(BND * 2);
  _Float16* k_h     = (_Float16*)allocb(BND * 2);
  _Float16* v_h     = (_Float16*)allocb(BND * 2);
  _Float16* vT_h    = (_Float16*)allocb(BND * 2);
  _Float16* o_h     = (_Float16*)allocb(BND * 2);
  _Float16* s_h     = (_Float16*)allocb((size_t)NPTS * NPTS * 2);
  _Float16* h_h     = (_Float16*)allocb((size_t)BNROWS * DFF * 2);
  _Float16* c1p  = (_Float16*)allocb(64 * 32 * 2);
  _Float16* c2p  = (_Float16*)allocb(64 * 64 * 2);
  _Float16* c3p  = (_Float16*)allocb(128 * 64 * 2);
  _Float16* c4p  = (_Float16*)allocb(256 * 128 * 2);
  _Float16* c5h  = (_Float16*)allocb(512 * 512 * 2);
  _Float16* wqh  = (_Float16*)allocb(512 * 512 * 2);
  _Float16* wkh  = (_Float16*)allocb(512 * 512 * 2);
  _Float16* wvh  = (_Float16*)allocb(512 * 512 * 2);
  _Float16* woh  = (_Float16*)allocb(512 * 512 * 2);
  _Float16* fw1h = (_Float16*)allocb((size_t)DFF * 512 * 2);
  _Float16* fw2h = (_Float16*)allocb((size_t)512 * DFF * 2);

  auto padcvt = [&](const float* w, int cout, int kact, int kp, _Float16* o) {
    int tot = cout * kp;
    padcvt_kernel<<<(tot + 255) / 256, 256, 0, stream>>>(w, cout, kact, kp, o);
  };
  padcvt(c1, 64, 26, 32, c1p);
  padcvt(c2, 64, 64, 64, c2p);
  padcvt(c3, 128, 64, 64, c3p);
  padcvt(c4, 256, 128, 128, c4p);
  padcvt(c5, 512, 512, 512, c5h);
  padcvt(wq, 512, 512, 512, wqh);
  padcvt(wk, 512, 512, 512, wkh);
  padcvt(wv, 512, 512, 512, wvh);
  padcvt(wo, 512, 512, 512, woh);
  padcvt(fw1, DFF, 512, 512, fw1h);
  padcvt(fw2, 512, DFF, DFF, fw2h);

  auto gemm = [&](const _Float16* A, int lda, const _Float16* Bm, int ldb,
                  const float* bias, void* C, int ldc, int M, int Nn, int Kd,
                  float scale, int relu, int c16) {
    dim3 g(Nn / 64, M / 128);
    gemm_kernel<<<g, 256, 0, stream>>>(A, lda, Bm, ldb, bias, C, ldc,
                                       Kd, scale, relu, c16);
  };

  // -------------------- DGCNN for both clouds --------------------
  auto dgcnn = [&](const float* pts, float* eT) {
    knn_kernel<<<BATCH * (NPTS / 128), 128, 0, stream>>>(pts, idxb);
    dgcnn_kernel<<<BATCH * NPTS, 64, 0, stream>>>(pts, idxb, c1p, c2p, c3p, c4p, cat_h);
    // conv5: eT[bn,o] = relu(sum_c cat[bn,c] * c5[o,c])
    gemm(cat_h, DMODEL, c5h, DMODEL, nullptr, eT, DMODEL,
         BNROWS, DMODEL, DMODEL, 1.0f, 1, 0);
  };
  dgcnn(src, eaT);
  dgcnn(tgt, ebT);

  // -------------------- cross attention --------------------
  const float invsq = 0.08838834764831845f;  // 1/sqrt(128)
  auto cross_attn = [&](const float* xT, const float* memT, float* outT) {
    ln_kernel<<<BNROWS, 256, 0, stream>>>(memT, lnma, lnmb, memln_h, 1);
    ln_kernel<<<BNROWS, 256, 0, stream>>>(xT, ln0a, ln0b, xln_h, 1);
    gemm(xln_h,   DMODEL, wqh, DMODEL, bq, q_h, DMODEL, BNROWS, DMODEL, DMODEL, 1.0f, 0, 1);
    gemm(memln_h, DMODEL, wkh, DMODEL, bk, k_h, DMODEL, BNROWS, DMODEL, DMODEL, 1.0f, 0, 1);
    gemm(memln_h, DMODEL, wvh, DMODEL, bv, v_h, DMODEL, BNROWS, DMODEL, DMODEL, 1.0f, 0, 1);
    vtrans_kernel<<<(unsigned)(BND / 256), 256, 0, stream>>>(v_h, vT_h);
    for (int b = 0; b < BATCH; ++b) {
      for (int h = 0; h < HEADS; ++h) {
        size_t oqk = (size_t)b * NPTS * DMODEL + (size_t)h * DKH;
        // scores = (Q K^T)/sqrt(dk)
        gemm(q_h + oqk, DMODEL, k_h + oqk, DMODEL, nullptr, scores, NPTS,
             NPTS, NPTS, DKH, invsq, 0, 0);
        softmax_kernel<<<NPTS, 256, 0, stream>>>(scores, s_h);
        // O_h = attn V_h   (V pre-transposed: vT[(b*D + d)][m])
        gemm(s_h, NPTS, vT_h + ((size_t)b * DMODEL + (size_t)h * DKH) * NPTS, NPTS,
             nullptr, o_h + oqk, DMODEL, NPTS, DKH, NPTS, 1.0f, 0, 1);
      }
    }
    gemm(o_h, DMODEL, woh, DMODEL, bo, projo, DMODEL, BNROWS, DMODEL, DMODEL, 1.0f, 0, 0);
    add_kernel<<<(unsigned)(BND / 256), 256, 0, stream>>>(xT, projo, x1f);   // x1
    ln_kernel<<<BNROWS, 256, 0, stream>>>(x1f, ln1a, ln1b, q_h, 1);          // ln1(x1) f16
    gemm(q_h, DMODEL, fw1h, DMODEL, fb1, hbuf, DFF, BNROWS, DFF, DMODEL, 1.0f, 1, 0);
    bn_stats_kernel<<<DFF, 256, 0, stream>>>(hbuf, mv);
    bn_apply_kernel<<<(unsigned)((size_t)BNROWS * DFF / 256), 256, 0, stream>>>(
        hbuf, fg, fbeta, mv, h_h);
    gemm(h_h, DFF, fw2h, DFF, fb2, projo, DMODEL, BNROWS, DMODEL, DFF, 1.0f, 0, 0);
    add_kernel<<<(unsigned)(BND / 256), 256, 0, stream>>>(x1f, projo, x2f);  // x2
    ln_kernel<<<BNROWS, 256, 0, stream>>>(x2f, lnoa, lnob, outT, 0);
  };
  cross_attn(eaT, ebT, paT);
  cross_attn(ebT, eaT, pbT);

  // -------------------- final: out = stack([ea+pa, eb+pb]) --------------------
  final_kernel<<<(unsigned)(BND / 256), 256, 0, stream>>>(eaT, paT, (float*)d_out);
  final_kernel<<<(unsigned)(BND / 256), 256, 0, stream>>>(ebT, pbT, (float*)d_out + BND);
}
